// GATModule_59390807769623
// MI455X (gfx1250) — compile-verified, matrix operand order checked by reading, not run on hardware
//
#include <hip/hip_runtime.h>
#include <hip/hip_bf16.h>

typedef __attribute__((ext_vector_type(16))) _Float16 v16h;
typedef __attribute__((ext_vector_type(8)))  _Float16 v8h;
typedef __attribute__((ext_vector_type(8)))  float    v8f;

#define N_NODES 50000
#define N_EDGES 800000
#define DIM     128
#define LDS_STRIDE 136   // f16 elements; 272B row pitch -> 16B aligned, bank-spread

// ---------------------------------------------------------------------------
// WMMA fragment layouts (wave32, V_WMMA_F32_16X16X32_F16, ISA 7.12.2)
// A (16x32, MxK): lane l holds row M = l&15; element j -> K = (j<8?0:16) + (l>>4)*8 + (j&7)
// B (32x16, KxN): lane l holds col N = l&15; same K mapping as A.
// C/D (16x16 f32): element i -> M = (l>>4)*8 + i, N = l&15.
// ---------------------------------------------------------------------------
__device__ __forceinline__ v16h load_a_frag_f32(const float* __restrict__ A,
                                                int row, int kbase, int half) {
    const float* p = A + row * DIM + kbase + half * 8;
    float4 x0 = *(const float4*)(p);
    float4 x1 = *(const float4*)(p + 4);
    float4 x2 = *(const float4*)(p + 16);
    float4 x3 = *(const float4*)(p + 20);
    v16h a;
    a[0] = (_Float16)x0.x;  a[1] = (_Float16)x0.y;  a[2] = (_Float16)x0.z;  a[3] = (_Float16)x0.w;
    a[4] = (_Float16)x1.x;  a[5] = (_Float16)x1.y;  a[6] = (_Float16)x1.z;  a[7] = (_Float16)x1.w;
    a[8] = (_Float16)x2.x;  a[9] = (_Float16)x2.y;  a[10] = (_Float16)x2.z; a[11] = (_Float16)x2.w;
    a[12] = (_Float16)x3.x; a[13] = (_Float16)x3.y; a[14] = (_Float16)x3.z; a[15] = (_Float16)x3.w;
    return a;
}

// B fragment from LDS-staged transposed f16 weights: Bt_lds[n*LDS_STRIDE + k]
__device__ __forceinline__ v16h load_b_frag_lds(const _Float16* p_lds,
                                                int ncol, int kbase, int half) {
    const _Float16* p = p_lds + ncol * LDS_STRIDE + kbase + half * 8;
    v8h lo = *(const v8h*)(p);        // K = kbase + half*8 + 0..7
    v8h hi = *(const v8h*)(p + 16);   // K = kbase + 16 + half*8 + 0..7
    v16h b;
#pragma unroll
    for (int j = 0; j < 8; ++j) { b[j] = lo[j]; b[j + 8] = hi[j]; }
    return b;
}

// ---------------------------------------------------------------------------
// One-time weight prep: Bt[n*128 + k] = (f16) W[k*128 + n]
// ---------------------------------------------------------------------------
__global__ void w_to_f16t(const float* __restrict__ W, _Float16* __restrict__ Bt) {
    int t = (int)(blockIdx.x * blockDim.x + threadIdx.x);
    if (t >= DIM * DIM) return;
    int n = t >> 7, k = t & 127;
    Bt[n * DIM + k] = (_Float16)W[k * DIM + n];
}

// ---------------------------------------------------------------------------
// C = act(A[Mx128] * W[128x128] + bias).  W supplied as transposed f16 Bt.
// Block = 128 threads (4 waves); stages Bt into LDS once, then each wave
// computes a 16-row tile with 32 WMMAs, B fragments via ds_load_b128.
// ---------------------------------------------------------------------------
__global__ void gemm128_wmma(const float* __restrict__ A, const _Float16* __restrict__ Bt,
                             const float* __restrict__ bias, float* __restrict__ C,
                             int M, int relu) {
    __shared__ _Float16 lb[DIM * LDS_STRIDE];   // 34,816 B

    // cooperative stage: 2048 16-byte chunks, 16 per thread
#pragma unroll
    for (int i = 0; i < 16; ++i) {
        int chunk = (int)threadIdx.x + i * 128;
        int n = chunk >> 4, c = chunk & 15;
        *(int4*)(lb + n * LDS_STRIDE + c * 8) = *(const int4*)(Bt + n * DIM + c * 8);
    }
    __syncthreads();

    int wave = (int)((blockIdx.x * blockDim.x + threadIdx.x) >> 5);
    int lane = (int)(threadIdx.x & 31);
    if (wave >= (M >> 4)) return;          // uniform per-wave branch (EXEC all-1s)
    int m0 = wave << 4;
    int r = lane & 15;
    int half = lane >> 4;

    v16h afrag[4];
#pragma unroll
    for (int ks = 0; ks < 4; ++ks)
        afrag[ks] = load_a_frag_f32(A, m0 + r, ks * 32, half);

#pragma unroll
    for (int nt = 0; nt < 8; ++nt) {
        v8f acc = {};
#pragma unroll
        for (int ks = 0; ks < 4; ++ks) {
            v16h bfrag = load_b_frag_lds(lb, nt * 16 + r, ks * 32, half);
            acc = __builtin_amdgcn_wmma_f32_16x16x32_f16(
                false, afrag[ks], false, bfrag, (short)0, acc, false, false);
        }
        int n = nt * 16 + r;
        float bv = bias[n];
#pragma unroll
        for (int i = 0; i < 8; ++i) {
            float v = acc[i] + bv;
            if (relu) v = fmaxf(v, 0.0f);
            C[(m0 + half * 8 + i) * DIM + n] = v;
        }
    }
}

// ---------------------------------------------------------------------------
// suv[N,8] = h @ [W_au | W_av | 0pad] (+ b_au on cols 0..3)
// ---------------------------------------------------------------------------
__global__ void scores_wmma(const float* __restrict__ h, const float* __restrict__ Wau,
                            const float* __restrict__ bau, const float* __restrict__ Wav,
                            float* __restrict__ suv, int M) {
    int wave = (int)((blockIdx.x * blockDim.x + threadIdx.x) >> 5);
    int lane = (int)(threadIdx.x & 31);
    if (wave >= (M >> 4)) return;
    int m0 = wave << 4;
    int r = lane & 15;
    int half = lane >> 4;

    v8f acc = {};
#pragma unroll
    for (int ks = 0; ks < 4; ++ks) {
        v16h a = load_a_frag_f32(h, m0 + r, ks * 32, half);
        v16h b;
#pragma unroll
        for (int j = 0; j < 16; ++j) {
            int K = ks * 32 + (j < 8 ? 0 : 16) + half * 8 + (j & 7);
            float w = 0.0f;
            if (r < 4)      w = Wau[K * 4 + r];
            else if (r < 8) w = Wav[K * 4 + (r - 4)];
            b[j] = (_Float16)w;
        }
        acc = __builtin_amdgcn_wmma_f32_16x16x32_f16(
            false, a, false, b, (short)0, acc, false, false);
    }
    if (r < 8) {
        float bv = (r < 4) ? bau[r] : 0.0f;
#pragma unroll
        for (int i = 0; i < 8; ++i)
            suv[(m0 + half * 8 + i) * 8 + r] = acc[i] + bv;
    }
}

// ---------------------------------------------------------------------------
// Edge softmax + aggregation
// ---------------------------------------------------------------------------
__device__ __forceinline__ unsigned enc_f32(float f) {
    unsigned u = __float_as_uint(f);
    return (u & 0x80000000u) ? ~u : (u | 0x80000000u);
}
__device__ __forceinline__ float dec_f32(unsigned u) {
    return __uint_as_float((u & 0x80000000u) ? (u & 0x7FFFFFFFu) : ~u);
}

__global__ void edge_pass1(const int* __restrict__ src, const int* __restrict__ dst,
                           const float* __restrict__ suv, float* __restrict__ ebuf,
                           unsigned* __restrict__ menc, int E) {
    int e = (int)(blockIdx.x * blockDim.x + threadIdx.x);
    if (e >= E) return;
    int s = src[e], d = dst[e];
    float4 su = *(const float4*)(suv + s * 8);
    float4 sv = *(const float4*)(suv + d * 8 + 4);
    float v[4] = { su.x + sv.x, su.y + sv.y, su.z + sv.z, su.w + sv.w };
    float lv[4];
#pragma unroll
    for (int j = 0; j < 4; ++j) {
        lv[j] = v[j] >= 0.0f ? v[j] : 0.2f * v[j];
        atomicMax(&menc[d * 4 + j], enc_f32(lv[j]));
    }
    *(float4*)(ebuf + e * 4) = make_float4(lv[0], lv[1], lv[2], lv[3]);
}

__global__ void edge_pass2(const int* __restrict__ dst, const unsigned* __restrict__ menc,
                           float* __restrict__ ebuf, float* __restrict__ z, int E) {
    int e = (int)(blockIdx.x * blockDim.x + threadIdx.x);
    if (e >= E) return;
    int d = dst[e];
    float4 ev = *(const float4*)(ebuf + e * 4);
    float in[4] = { ev.x, ev.y, ev.z, ev.w };
    float out[4];
#pragma unroll
    for (int j = 0; j < 4; ++j) {
        float m = dec_f32(menc[d * 4 + j]);
        out[j] = __expf(in[j] - m);
        atomicAdd(&z[d * 4 + j], out[j]);
    }
    *(float4*)(ebuf + e * 4) = make_float4(out[0], out[1], out[2], out[3]);
}

// agg[dst, d] += h[src, d] * p[d & 3]   (32 lanes / edge, float4 per lane)
__global__ void edge_agg(const int* __restrict__ src, const int* __restrict__ dst,
                         const float* __restrict__ ex, const float* __restrict__ z,
                         const float* __restrict__ h, float* __restrict__ agg, int E) {
    long gid = (long)blockIdx.x * blockDim.x + threadIdx.x;
    int e = (int)(gid >> 5);
    int lane = (int)(gid & 31);
    if (e >= E) return;
    int s = src[e], d = dst[e];
    float4 exv = *(const float4*)(ex + e * 4);
    const float* zp = z + d * 4;
    float p0 = exv.x / zp[0];
    float p1 = exv.y / zp[1];
    float p2 = exv.z / zp[2];
    float p3 = exv.w / zp[3];
    float4 hv = *(const float4*)(h + s * DIM + lane * 4);
    float* ap = agg + d * DIM + lane * 4;
    atomicAdd(ap + 0, hv.x * p0);
    atomicAdd(ap + 1, hv.y * p1);
    atomicAdd(ap + 2, hv.z * p2);
    atomicAdd(ap + 3, hv.w * p3);
}

__global__ void zero_u32(unsigned* __restrict__ p, int n) {
    int i = (int)(blockIdx.x * blockDim.x + threadIdx.x);
    if (i < n) p[i] = 0u;
}

// ---------------------------------------------------------------------------
extern "C" void kernel_launch(void* const* d_in, const int* in_sizes, int n_in,
                              void* d_out, int out_size, void* d_ws, size_t ws_size,
                              hipStream_t stream) {
    const float* x    = (const float*)d_in[0];
    const int*   src  = (const int*)  d_in[1];
    const int*   dst  = (const int*)  d_in[2];
    const float* W_in = (const float*)d_in[3];
    const float* b_in = (const float*)d_in[4];
    const float* W_au = (const float*)d_in[5];
    const float* b_au = (const float*)d_in[6];
    const float* W_av = (const float*)d_in[7];
    const float* W_f1 = (const float*)d_in[8];
    const float* b_f1 = (const float*)d_in[9];
    const float* W_f2 = (const float*)d_in[10];
    const float* b_f2 = (const float*)d_in[11];
    float* out = (float*)d_out;

    // workspace layout
    float* ws = (float*)d_ws;
    float*    h    = ws;                                      // N*128  (reused as t)
    float*    suv  = h + (size_t)N_NODES * DIM;               // N*8
    float*    ebuf = suv + (size_t)N_NODES * 8;               // E*4 (e, then ex in place)
    unsigned* menc = (unsigned*)(ebuf + (size_t)N_EDGES * 4); // N*4
    float*    z    = (float*)(menc + (size_t)N_NODES * 4);    // N*4
    float*    agg  = z + (size_t)N_NODES * 4;                 // N*128
    _Float16* wtin = (_Float16*)(agg + (size_t)N_NODES * DIM);// 128*128 f16
    _Float16* wtf1 = wtin + (size_t)DIM * DIM;                // 128*128 f16
    _Float16* wtf2 = wtf1 + (size_t)DIM * DIM;                // 128*128 f16

    const int BLK = 256;
    // zero segment-max encodings (encoded 0 == -inf sentinel), z, agg
    zero_u32<<<(N_NODES * 4 + BLK - 1) / BLK, BLK, 0, stream>>>(menc, N_NODES * 4);
    zero_u32<<<(N_NODES * 4 + BLK - 1) / BLK, BLK, 0, stream>>>((unsigned*)z, N_NODES * 4);
    zero_u32<<<(N_NODES * DIM + BLK - 1) / BLK, BLK, 0, stream>>>((unsigned*)agg, N_NODES * DIM);

    // weight prep: transposed f16 copies (done once per launch; trivially cheap)
    const int wgrd = (DIM * DIM + BLK - 1) / BLK;
    w_to_f16t<<<wgrd, BLK, 0, stream>>>(W_in, wtin);
    w_to_f16t<<<wgrd, BLK, 0, stream>>>(W_f1, wtf1);
    w_to_f16t<<<wgrd, BLK, 0, stream>>>(W_f2, wtf2);

    // GEMM grid: one wave per 16-row tile, 4 waves per 128-thread block
    const int tiles = N_NODES / 16;                 // 3125 (exact: 50000 % 16 == 0)
    const int gblk = 128;
    const int ggrd = (tiles + 3) / 4;

    // h = x @ W_in + b_in
    gemm128_wmma<<<ggrd, gblk, 0, stream>>>(x, wtin, b_in, h, N_NODES, 0);
    // suv = [h@W_au + b_au | h@W_av]
    scores_wmma<<<ggrd, gblk, 0, stream>>>(h, W_au, b_au, W_av, suv, N_NODES);

    // edge softmax
    edge_pass1<<<(N_EDGES + BLK - 1) / BLK, BLK, 0, stream>>>(src, dst, suv, ebuf, menc, N_EDGES);
    edge_pass2<<<(N_EDGES + BLK - 1) / BLK, BLK, 0, stream>>>(dst, menc, ebuf, z, N_EDGES);

    // weighted aggregation: 32 lanes per edge
    long aggThreads = (long)N_EDGES * 32;
    edge_agg<<<(int)((aggThreads + BLK - 1) / BLK), BLK, 0, stream>>>(src, dst, ebuf, z, h, agg, N_EDGES);

    // FFN: t = relu(agg @ W_f1 + b_f1)  (t reuses h buffer); out = t @ W_f2 + b_f2
    gemm128_wmma<<<ggrd, gblk, 0, stream>>>(agg, wtf1, b_f1, h, N_NODES, 1);
    gemm128_wmma<<<ggrd, gblk, 0, stream>>>(h, wtf2, b_f2, out, N_NODES, 0);
}
